// MultiHeadAttention_44573170598490
// MI455X (gfx1250) — matrix operand order repaired and checked by reading, not compiled, not run on hardware
//
#include <hip/hip_runtime.h>
#include <hip/hip_bf16.h>
#include <math.h>

typedef __attribute__((ext_vector_type(16))) _Float16 v16h;
typedef __attribute__((ext_vector_type(8)))  _Float16 v8h;
typedef __attribute__((ext_vector_type(8)))  float    v8f;

static constexpr int NB  = 4;
static constexpr int SEQ = 2048;
static constexpr int DM  = 1024;
static constexpr int NH  = 16;
static constexpr int HDIM = 64;
#define ATTN_SCALE 0.125f

// GEMM tiling
static constexpr int BM = 128;
static constexpr int BN = 64;
static constexpr int BK = 32;
static constexpr int LDS_STRIDE = 48;   // halfs; 96B rows keep 16B alignment, avoids bank conflicts

__device__ __forceinline__ v16h load_frag(const _Float16* p0, const _Float16* p1) {
    union { v16h v; v8h h[2]; } u;
    u.h[0] = *(const v8h*)p0;
    u.h[1] = *(const v8h*)p1;
    return u.v;
}

// ---------------------------------------------------------------------------
// Fused Q/K/V projection: Y = X @ W + b, fp32 in, f16 out.
// z=0 -> Qh[b][h][s][hd], z=1 -> Kh[b][h][s][hd], z=2 -> Vt[b][h][hd][s]
// ---------------------------------------------------------------------------
__global__ __launch_bounds__(256) void qkv_proj_kernel(
    const float* __restrict__ xq, const float* __restrict__ xk, const float* __restrict__ xv,
    const float* __restrict__ Wq, const float* __restrict__ bq,
    const float* __restrict__ Wk, const float* __restrict__ bk,
    const float* __restrict__ Wv, const float* __restrict__ bv,
    _Float16* __restrict__ Qh, _Float16* __restrict__ Kh, _Float16* __restrict__ Vt)
{
    __shared__ alignas(16) _Float16 sA[BM * LDS_STRIDE];
    __shared__ alignas(16) _Float16 sB[BN * LDS_STRIDE];

    const int tid = threadIdx.x;
    const int z   = blockIdx.z;
    const float* X    = (z == 0) ? xq : (z == 1) ? xk : xv;
    const float* W    = (z == 0) ? Wq : (z == 1) ? Wk : Wv;
    const float* bias = (z == 0) ? bq : (z == 1) ? bk : bv;

    const int m0   = blockIdx.x * BM;
    const int n0   = blockIdx.y * BN;
    const int lane = tid & 31;
    const int wid  = tid >> 5;
    const int wm   = (wid & 3) * 32;   // 4 waves along M
    const int wn   = (wid >> 2) * 32;  // 2 waves along N
    const int fm   = lane & 15;
    const int hi   = lane >> 4;

    v8f acc[2][2] = {};

    for (int kk = 0; kk < DM / BK; ++kk) {
        const int kbase = kk * BK;
        __syncthreads();
        // Stage A (BMxBK fp32 -> f16), 4 float4 per thread
        #pragma unroll
        for (int it = 0; it < 4; ++it) {
            int idx = tid + 256 * it;            // 0..1023
            int row = idx >> 3;
            int qd  = (idx & 7) << 2;
            const float4 xv4 = *reinterpret_cast<const float4*>(
                X + (size_t)(m0 + row) * DM + kbase + qd);
            _Float16* d = &sA[row * LDS_STRIDE + qd];
            d[0] = (_Float16)xv4.x; d[1] = (_Float16)xv4.y;
            d[2] = (_Float16)xv4.z; d[3] = (_Float16)xv4.w;
        }
        // Stage B transposed (W is [K][N] row-major -> sB is [N][K])
        #pragma unroll
        for (int it = 0; it < 2; ++it) {
            int idx = tid + 256 * it;            // 0..511
            int kr  = idx >> 4;
            int nq  = (idx & 15) << 2;
            const float4 wv4 = *reinterpret_cast<const float4*>(
                W + (size_t)(kbase + kr) * DM + n0 + nq);
            sB[(nq + 0) * LDS_STRIDE + kr] = (_Float16)wv4.x;
            sB[(nq + 1) * LDS_STRIDE + kr] = (_Float16)wv4.y;
            sB[(nq + 2) * LDS_STRIDE + kr] = (_Float16)wv4.z;
            sB[(nq + 3) * LDS_STRIDE + kr] = (_Float16)wv4.w;
        }
        __syncthreads();

        v16h af[2], bf[2];
        #pragma unroll
        for (int mi = 0; mi < 2; ++mi) {
            const _Float16* p = &sA[(wm + mi * 16 + fm) * LDS_STRIDE + hi * 8];
            af[mi] = load_frag(p, p + 16);       // A layout: K chunks {0..7,16..23}/{8..15,24..31}
        }
        #pragma unroll
        for (int ni = 0; ni < 2; ++ni) {
            const _Float16* p = &sB[(wn + ni * 16 + fm) * LDS_STRIDE + hi * 8];
            bf[ni] = load_frag(p, p + 16);       // B layout: K chunks {0..7,16..23}/{8..15,24..31}
        }
        #pragma unroll
        for (int mi = 0; mi < 2; ++mi)
            #pragma unroll
            for (int ni = 0; ni < 2; ++ni)
                acc[mi][ni] = __builtin_amdgcn_wmma_f32_16x16x32_f16(
                    false, af[mi], false, bf[ni], (short)0, acc[mi][ni], false, false);
    }

    // Epilogue: bias + f16 store into head-split layouts
    #pragma unroll
    for (int mi = 0; mi < 2; ++mi) {
        #pragma unroll
        for (int ni = 0; ni < 2; ++ni) {
            const int n = n0 + wn + ni * 16 + fm;
            const float bval = bias[n];
            const int h = n >> 6, hd = n & 63;
            #pragma unroll
            for (int e = 0; e < 8; ++e) {
                const int m = m0 + wm + mi * 16 + e + 8 * hi;
                const int b = m >> 11;            // / SEQ
                const int s = m & (SEQ - 1);
                const float val = acc[mi][ni][e] + bval;
                if (z == 2) {
                    Vt[(((size_t)b * NH + h) * HDIM + hd) * SEQ + s] = (_Float16)val;
                } else {
                    _Float16* dst = (z == 0) ? Qh : Kh;
                    dst[(((size_t)b * NH + h) * SEQ + s) * HDIM + hd] = (_Float16)val;
                }
            }
        }
    }
}

// ---------------------------------------------------------------------------
// Flash attention: 1 wave = 16 queries of one (b,h); online softmax; all
// matmuls on v_wmma_f32_16x16x32_f16. Writes f16 O in [B*S][D] layout.
// ---------------------------------------------------------------------------
__global__ __launch_bounds__(256) void attn_kernel(
    const _Float16* __restrict__ Qh, const _Float16* __restrict__ Kh,
    const _Float16* __restrict__ Vt, _Float16* __restrict__ O16)
{
    __shared__ alignas(16) _Float16 sP[8 * 16 * 32];   // per-wave 16x32 P tile

    const int tid  = threadIdx.x;
    const int lane = tid & 31;
    const int w    = tid >> 5;
    const int fm   = lane & 15;
    const int hi   = lane >> 4;
    const int bh   = blockIdx.y;
    const int b    = bh >> 4;       // / NH
    const int h    = bh & 15;
    const _Float16* Qb = Qh + (size_t)bh * SEQ * HDIM;
    const _Float16* Kb = Kh + (size_t)bh * SEQ * HDIM;
    const _Float16* Vb = Vt + (size_t)bh * HDIM * SEQ;
    _Float16* sPw = &sP[w * 16 * 32];
    const int q0 = blockIdx.x * 128 + w * 16;

    // Q fragments (A layout), resident for the whole key loop
    v16h qf[2];
    #pragma unroll
    for (int kc = 0; kc < 2; ++kc) {
        const _Float16* p = Qb + (size_t)(q0 + fm) * HDIM + kc * 32 + hi * 8;
        qf[kc] = load_frag(p, p + 16);
    }

    v8f oacc[4] = {};
    float rm[8], rl[8];
    #pragma unroll
    for (int e = 0; e < 8; ++e) { rm[e] = -3.0e38f; rl[e] = 0.f; }

    for (int j = 0; j < SEQ / 32; ++j) {
        const int k0 = j * 32;
        // scores: 16 q x 32 keys
        v8f s0 = {}, s1 = {};
        #pragma unroll
        for (int kc = 0; kc < 2; ++kc) {
            const _Float16* pk0 = Kb + (size_t)(k0 + fm) * HDIM + kc * 32 + hi * 8;
            const _Float16* pk1 = Kb + (size_t)(k0 + 16 + fm) * HDIM + kc * 32 + hi * 8;
            v16h b0 = load_frag(pk0, pk0 + 16);
            v16h b1 = load_frag(pk1, pk1 + 16);
            s0 = __builtin_amdgcn_wmma_f32_16x16x32_f16(false, qf[kc], false, b0, (short)0, s0, false, false);
            s1 = __builtin_amdgcn_wmma_f32_16x16x32_f16(false, qf[kc], false, b1, (short)0, s1, false, false);
        }
        // online softmax; row e+8*hi lives across the 16 lanes of this half-wave
        #pragma unroll
        for (int e = 0; e < 8; ++e) {
            float a0 = s0[e] * ATTN_SCALE;
            float a1 = s1[e] * ATTN_SCALE;
            float mt = fmaxf(a0, a1);
            mt = fmaxf(mt, __shfl_xor(mt, 1));
            mt = fmaxf(mt, __shfl_xor(mt, 2));
            mt = fmaxf(mt, __shfl_xor(mt, 4));
            mt = fmaxf(mt, __shfl_xor(mt, 8));
            float mn    = fmaxf(rm[e], mt);
            float alpha = __expf(rm[e] - mn);
            float p0 = __expf(a0 - mn);
            float p1 = __expf(a1 - mn);
            float ps = p0 + p1;
            ps += __shfl_xor(ps, 1);
            ps += __shfl_xor(ps, 2);
            ps += __shfl_xor(ps, 4);
            ps += __shfl_xor(ps, 8);
            rl[e] = rl[e] * alpha + ps;
            rm[e] = mn;
            #pragma unroll
            for (int c = 0; c < 4; ++c) oacc[c][e] *= alpha;
            const int rrow = e + 8 * hi;
            sPw[rrow * 32 + fm]      = (_Float16)p0;   // C-layout -> row-major P in LDS
            sPw[rrow * 32 + 16 + fm] = (_Float16)p1;
        }
        // P back as A fragment (same-wave LDS ops are in-order)
        const _Float16* pp = &sPw[fm * 32 + hi * 8];
        v16h pf = load_frag(pp, pp + 16);
        // O += P @ V  (V stored transposed: contiguous keys per hd column)
        #pragma unroll
        for (int c = 0; c < 4; ++c) {
            const _Float16* pv = Vb + (size_t)(c * 16 + fm) * SEQ + k0 + hi * 8;
            v16h vf = load_frag(pv, pv + 16);
            oacc[c] = __builtin_amdgcn_wmma_f32_16x16x32_f16(false, pf, false, vf, (short)0, oacc[c], false, false);
        }
    }

    #pragma unroll
    for (int c = 0; c < 4; ++c) {
        const int col = h * HDIM + c * 16 + fm;
        #pragma unroll
        for (int e = 0; e < 8; ++e) {
            const int s = q0 + e + 8 * hi;
            const float val = oacc[c][e] / rl[e];
            O16[((size_t)b * SEQ + s) * DM + col] = (_Float16)val;
        }
    }
}

// ---------------------------------------------------------------------------
// Output projection: out = O16 @ Wo + bo, f16 A, fp32 result.
// ---------------------------------------------------------------------------
__global__ __launch_bounds__(256) void out_proj_kernel(
    const _Float16* __restrict__ A, const float* __restrict__ Wo,
    const float* __restrict__ bo, float* __restrict__ out)
{
    __shared__ alignas(16) _Float16 sA[BM * LDS_STRIDE];
    __shared__ alignas(16) _Float16 sB[BN * LDS_STRIDE];

    const int tid  = threadIdx.x;
    const int m0   = blockIdx.x * BM;
    const int n0   = blockIdx.y * BN;
    const int lane = tid & 31;
    const int wid  = tid >> 5;
    const int wm   = (wid & 3) * 32;
    const int wn   = (wid >> 2) * 32;
    const int fm   = lane & 15;
    const int hi   = lane >> 4;

    v8f acc[2][2] = {};

    for (int kk = 0; kk < DM / BK; ++kk) {
        const int kbase = kk * BK;
        __syncthreads();
        // Stage A (already f16): 2 x 16B per thread
        #pragma unroll
        for (int it = 0; it < 2; ++it) {
            int idx = tid + 256 * it;            // 0..511
            int row = idx >> 2;
            int oq  = (idx & 3) << 3;
            *(v8h*)&sA[row * LDS_STRIDE + oq] =
                *(const v8h*)(A + (size_t)(m0 + row) * DM + kbase + oq);
        }
        // Stage B transposed, fp32 -> f16
        #pragma unroll
        for (int it = 0; it < 2; ++it) {
            int idx = tid + 256 * it;
            int kr  = idx >> 4;
            int nq  = (idx & 15) << 2;
            const float4 wv4 = *reinterpret_cast<const float4*>(
                Wo + (size_t)(kbase + kr) * DM + n0 + nq);
            sB[(nq + 0) * LDS_STRIDE + kr] = (_Float16)wv4.x;
            sB[(nq + 1) * LDS_STRIDE + kr] = (_Float16)wv4.y;
            sB[(nq + 2) * LDS_STRIDE + kr] = (_Float16)wv4.z;
            sB[(nq + 3) * LDS_STRIDE + kr] = (_Float16)wv4.w;
        }
        __syncthreads();

        v16h af[2], bf[2];
        #pragma unroll
        for (int mi = 0; mi < 2; ++mi) {
            const _Float16* p = &sA[(wm + mi * 16 + fm) * LDS_STRIDE + hi * 8];
            af[mi] = load_frag(p, p + 16);
        }
        #pragma unroll
        for (int ni = 0; ni < 2; ++ni) {
            const _Float16* p = &sB[(wn + ni * 16 + fm) * LDS_STRIDE + hi * 8];
            bf[ni] = load_frag(p, p + 16);
        }
        #pragma unroll
        for (int mi = 0; mi < 2; ++mi)
            #pragma unroll
            for (int ni = 0; ni < 2; ++ni)
                acc[mi][ni] = __builtin_amdgcn_wmma_f32_16x16x32_f16(
                    false, af[mi], false, bf[ni], (short)0, acc[mi][ni], false, false);
    }

    #pragma unroll
    for (int mi = 0; mi < 2; ++mi) {
        #pragma unroll
        for (int ni = 0; ni < 2; ++ni) {
            const int n = n0 + wn + ni * 16 + fm;
            const float bval = bo[n];
            #pragma unroll
            for (int e = 0; e < 8; ++e) {
                const int m = m0 + wm + mi * 16 + e + 8 * hi;
                out[(size_t)m * DM + n] = acc[mi][ni][e] + bval;
            }
        }
    }
}

extern "C" void kernel_launch(void* const* d_in, const int* in_sizes, int n_in,
                              void* d_out, int out_size, void* d_ws, size_t ws_size,
                              hipStream_t stream) {
    const float* q  = (const float*)d_in[0];
    const float* k  = (const float*)d_in[1];
    const float* v  = (const float*)d_in[2];
    const float* Wq = (const float*)d_in[3];
    const float* bq = (const float*)d_in[4];
    const float* Wk = (const float*)d_in[5];
    const float* bk = (const float*)d_in[6];
    const float* Wv = (const float*)d_in[7];
    const float* bv = (const float*)d_in[8];
    const float* Wo = (const float*)d_in[9];
    const float* bo = (const float*)d_in[10];
    float* out = (float*)d_out;

    _Float16* ws = (_Float16*)d_ws;
    const size_t HE = (size_t)NB * NH * SEQ * HDIM;   // 8,388,608 halfs = 16 MB
    _Float16* Qh  = ws;
    _Float16* Kh  = ws + HE;
    _Float16* Vt  = ws + 2 * HE;
    _Float16* O16 = ws + 3 * HE;

    // Q/K/V projections (z selects which)
    qkv_proj_kernel<<<dim3(64, 16, 3), 256, 0, stream>>>(
        q, k, v, Wq, bq, Wk, bk, Wv, bv, Qh, Kh, Vt);
    // flash attention per (128-query block, b*h)
    attn_kernel<<<dim3(16, 64), 256, 0, stream>>>(Qh, Kh, Vt, O16);
    // output projection
    out_proj_kernel<<<dim3(64, 16), 256, 0, stream>>>(O16, Wo, bo, out);

    (void)in_sizes; (void)n_in; (void)out_size; (void)ws_size;
}
